// LSTMCell_57303453663323
// MI455X (gfx1250) — matrix operand order repaired and checked by reading, not compiled, not run on hardware
//
#include <hip/hip_runtime.h>

typedef __attribute__((ext_vector_type(16))) __bf16 v16bf;
typedef __attribute__((ext_vector_type(8)))  __bf16 v8bf;
typedef __attribute__((ext_vector_type(8)))  float  v8f;
typedef unsigned short u16;
typedef unsigned int   u32;

#define B_DIM 4096
#define H_DIM 2048
#define K_DIM 4096   // H + I
#define NGATE 4
#define LDS_STRIDE 40   // 32 halves + 8 pad -> 80B rows (16B aligned, bank-spread)

__device__ __forceinline__ u16 f32_to_bf16(float f) {
    u32 u = __float_as_uint(f);
    u += 0x7FFFu + ((u >> 16) & 1u);   // round-to-nearest-even
    return (u16)(u >> 16);
}

__device__ __forceinline__ float fast_sigmoid(float x) {
    return 1.0f / (1.0f + __expf(-x));
}
__device__ __forceinline__ float fast_tanh(float x) {
    float e2 = __expf(2.0f * x);
    return 1.0f - 2.0f / (e2 + 1.0f);
}

// ---- pass 1: A_bf16[b][k] = bf16(concat(prev_state, x)[b][k]) --------------
__global__ __launch_bounds__(256)
void cvt_A_kernel(const float* __restrict__ prev, const float* __restrict__ x,
                  u16* __restrict__ A) {
    long idx = ((long)blockIdx.x * 256 + threadIdx.x) * 8;       // 8 elems/thread
    int b = (int)(idx >> 12);
    int k = (int)(idx & (K_DIM - 1));
    const float* src = (k < H_DIM) ? (prev + (long)b * H_DIM + k)
                                   : (x    + (long)b * H_DIM + (k - H_DIM));
    float4 f0 = reinterpret_cast<const float4*>(src)[0];
    float4 f1 = reinterpret_cast<const float4*>(src)[1];
    u16 r[8] = { f32_to_bf16(f0.x), f32_to_bf16(f0.y), f32_to_bf16(f0.z), f32_to_bf16(f0.w),
                 f32_to_bf16(f1.x), f32_to_bf16(f1.y), f32_to_bf16(f1.z), f32_to_bf16(f1.w) };
    *reinterpret_cast<uint4*>(A + idx) = *reinterpret_cast<const uint4*>(r);
}

// ---- pass 2: Wb[g*H + h][k] = bf16(W_g[h][k]) ------------------------------
__global__ __launch_bounds__(256)
void cvt_W_kernel(const float* __restrict__ Wfg, const float* __restrict__ Wig,
                  const float* __restrict__ Wog, const float* __restrict__ Wol,
                  u16* __restrict__ Wb) {
    long idx = ((long)blockIdx.x * 256 + threadIdx.x) * 8;
    int  g   = (int)(idx >> 23);                  // H*K = 2^23 elements per gate
    long off = idx & ((1L << 23) - 1);
    const float* src = (g == 0) ? Wfg : (g == 1) ? Wig : (g == 2) ? Wog : Wol;
    src += off;
    float4 f0 = reinterpret_cast<const float4*>(src)[0];
    float4 f1 = reinterpret_cast<const float4*>(src)[1];
    u16 r[8] = { f32_to_bf16(f0.x), f32_to_bf16(f0.y), f32_to_bf16(f0.z), f32_to_bf16(f0.w),
                 f32_to_bf16(f1.x), f32_to_bf16(f1.y), f32_to_bf16(f1.z), f32_to_bf16(f1.w) };
    *reinterpret_cast<uint4*>(Wb + idx) = *reinterpret_cast<const uint4*>(r);
}

// ---- CDNA5 async global->LDS (ASYNCcnt-tracked DMA, no VGPR round trip) ----
__device__ __forceinline__ void async_ld_b128(unsigned lds_off, const void* gptr) {
    asm volatile("global_load_async_to_lds_b128 %0, %1, off"
                 :: "v"(lds_off), "v"(gptr) : "memory");
}

// ---- fused 4-gate GEMM + LSTM epilogue -------------------------------------
// Block: 256 threads = 8 waves in a 4(M) x 2(N) grid.
// Block tile: 128 rows (b) x 64 cols (h); each wave: 32x32 per gate.
// Double-buffered LDS, async-DMA software pipeline (6 asyncs/wave/tile).
__global__ __launch_bounds__(256)
void lstm_wmma_kernel(const u16* __restrict__ A, const u16* __restrict__ W,
                      const float* __restrict__ c_state, float* __restrict__ out) {
    const int tid   = threadIdx.x;
    const int lane  = tid & 31;
    const int wid   = tid >> 5;
    const int waveM = wid >> 1;          // 0..3
    const int waveN = wid & 1;           // 0..1
    const int blockM = blockIdx.y * 128;
    const int blockN = blockIdx.x * 64;

    __shared__ u16 Alds[2][128 * LDS_STRIDE];
    __shared__ u16 Wlds[2][NGATE][64 * LDS_STRIDE];

    v8f acc[NGATE][2][2] = {};

    const int am0 = waveM * 32 + (lane & 15);
    const int ak0 = (lane < 16) ? 0 : 8;     // A: K 0-7/16-23 vs 8-15/24-31
    const int wk0 = (lane < 16) ? 0 : 8;     // B: K 0-7/16-23 vs 8-15/24-31

    // per-thread staging chunk: 16B, lane-contiguous within each async op
    const int sr = tid >> 2;               // 0..63
    const int sc = (tid & 3) * 8;          // 0,8,16,24 halves

    // ---- stage tile at K-offset kk into buffer `buf` (6 async ops / wave) --
    auto stage = [&](int buf, int kk) {
        #pragma unroll
        for (int j = 0; j < 2; ++j) {      // A: 128x32 halves = 8KB
            int row = j * 64 + sr;
            const u16* gp = A + (long)(blockM + row) * K_DIM + kk + sc;
            async_ld_b128((unsigned)(unsigned long long)
                              &Alds[buf][row * LDS_STRIDE + sc], gp);
        }
        #pragma unroll
        for (int g = 0; g < NGATE; ++g) {  // W: 4x64x32 halves = 16KB
            const u16* gp = W + (long)(g * H_DIM + blockN + sr) * K_DIM + kk + sc;
            async_ld_b128((unsigned)(unsigned long long)
                              &Wlds[buf][g][sr * LDS_STRIDE + sc], gp);
        }
    };

    stage(0, 0);
    int buf = 0;

    for (int kk = 0; kk < K_DIM; kk += 32) {
        // issue DMA for next tile into the other buffer (clamped: uniform, branchless)
        int knext = (kk + 32 < K_DIM) ? (kk + 32) : 0;
        stage(buf ^ 1, knext);

        // previous 6 asyncs (current tile) complete in order -> wait <=6
        asm volatile("s_wait_asynccnt 0x6" ::: "memory");
        __syncthreads();

        // ---- load ALL fragments first, then 16 back-to-back WMMAs ----
        v16bf afr[2];
        #pragma unroll
        for (int i = 0; i < 2; ++i) {
            const u16* base = &Alds[buf][(am0 + i * 16) * LDS_STRIDE];
            union { v16bf v; v8bf h[2]; } u;
            u.h[0] = *reinterpret_cast<const v8bf*>(base + ak0);
            u.h[1] = *reinterpret_cast<const v8bf*>(base + ak0 + 16);
            afr[i] = u.v;
        }
        v16bf bfr[NGATE][2];
        #pragma unroll
        for (int g = 0; g < NGATE; ++g) {
            #pragma unroll
            for (int j = 0; j < 2; ++j) {
                const u16* base =
                    &Wlds[buf][g][(waveN * 32 + j * 16 + (lane & 15)) * LDS_STRIDE];
                union { v16bf v; v8bf h[2]; } u;
                u.h[0] = *reinterpret_cast<const v8bf*>(base + wk0);
                u.h[1] = *reinterpret_cast<const v8bf*>(base + wk0 + 16);
                bfr[g][j] = u.v;
            }
        }
        #pragma unroll
        for (int g = 0; g < NGATE; ++g)
            #pragma unroll
            for (int j = 0; j < 2; ++j)
                #pragma unroll
                for (int i = 0; i < 2; ++i)
                    acc[g][i][j] = __builtin_amdgcn_wmma_f32_16x16x32_bf16(
                        false, afr[i], false, bfr[g][j], (short)0, acc[g][i][j],
                        false, false);

        __syncthreads();   // all waves done reading `buf` before it is re-filled
        buf ^= 1;
    }

    // ---- fused LSTM epilogue: C/D layout: vgpr e, lane l -> m = e + (l>=16)*8, n = l&15
    #pragma unroll
    for (int i = 0; i < 2; ++i) {
        #pragma unroll
        for (int j = 0; j < 2; ++j) {
            int mbase = blockM + waveM * 32 + i * 16 + ((lane >= 16) ? 8 : 0);
            int h     = blockN + waveN * 32 + j * 16 + (lane & 15);
            #pragma unroll
            for (int e = 0; e < 8; ++e) {
                int   b   = mbase + e;
                float ft  = fast_sigmoid(acc[0][i][j][e]);
                float it  = fast_sigmoid(acc[1][i][j][e]);
                float ctt = fast_sigmoid(acc[2][i][j][e]);   // NOTE: sigmoid per ref
                float ot  = fast_sigmoid(acc[3][i][j][e]);
                float cs  = c_state[(long)b * H_DIM + h];
                float ct  = ft * cs + it * ctt;
                out[(long)b * H_DIM + h] = ot * fast_tanh(ct);
            }
        }
    }
}

extern "C" void kernel_launch(void* const* d_in, const int* in_sizes, int n_in,
                              void* d_out, int out_size, void* d_ws, size_t ws_size,
                              hipStream_t stream) {
    const float* x          = (const float*)d_in[0];
    const float* prev_state = (const float*)d_in[1];
    const float* c_state    = (const float*)d_in[2];
    const float* W_fg       = (const float*)d_in[3];
    const float* W_ig       = (const float*)d_in[4];
    const float* W_og       = (const float*)d_in[5];
    const float* W_ol       = (const float*)d_in[6];
    float*       out        = (float*)d_out;

    u16* A_bf = (u16*)d_ws;                                        // 32 MB
    u16* W_bf = (u16*)((char*)d_ws + (size_t)B_DIM * K_DIM * 2);   // +64 MB

    cvt_A_kernel<<<(B_DIM * (long)K_DIM / 8) / 256, 256, 0, stream>>>(
        prev_state, x, A_bf);
    cvt_W_kernel<<<((long)NGATE * H_DIM * K_DIM / 8) / 256, 256, 0, stream>>>(
        W_fg, W_ig, W_og, W_ol, W_bf);

    dim3 grid(H_DIM / 64, B_DIM / 128);   // (32, 32)
    lstm_wmma_kernel<<<grid, 256, 0, stream>>>(A_bf, W_bf, c_state, out);
}
